// EnhancedGNNTransformerEncoder_82815559402266
// MI455X (gfx1250) — compile-verified, matrix operand order checked by reading, not run on hardware
//
#include <hip/hip_runtime.h>
#include <hip/hip_bf16.h>

// ---------------------------------------------------------------------------
// Types / helpers
// ---------------------------------------------------------------------------
typedef __attribute__((ext_vector_type(16))) __bf16 v16bf;
typedef __attribute__((ext_vector_type(8)))  __bf16 v8bf;
typedef __attribute__((ext_vector_type(8)))  float  v8f;

#define CAT8_8(lo, hi) __builtin_shufflevector((lo), (hi), 0,1,2,3,4,5,6,7,8,9,10,11,12,13,14,15)

static __device__ __forceinline__ v8f wmma_bf16(v16bf a, v16bf b, v8f c) {
  // D = A(16x32) * B(32x16) + C, f32 accumulate
  return __builtin_amdgcn_wmma_f32_16x16x32_bf16(false, a, false, b, (short)0, c, false, false);
}

#define NN      16384
#define HDIM    512
#define OUTC    256
#define NG      32
#define NPG     512
#define DEG     16

// ---------------------------------------------------------------------------
// f32 -> bf16 elementwise convert
// ---------------------------------------------------------------------------
__global__ __launch_bounds__(256) void cvt_f32_bf16(const float* __restrict__ src,
                                                    __bf16* __restrict__ dst, int n) {
  int i = blockIdx.x * 256 + threadIdx.x;
  if (i < n) dst[i] = (__bf16)src[i];
}

// ---------------------------------------------------------------------------
// Pack f32 weight into CDNA5 WMMA B-fragment layout (bf16).
// Logical B is [K, Ntot] (GEMM contraction K, output col n).
//   transpose==0: W is row-major [K, Nsub],  B[k][n] = W[k*Nsub + n]
//   transpose==1: W is row-major [Nsub, K],  B[k][n] = W[n*K + k]
// Packed block for (nt, kt): 32 lanes x 16 bf16, contiguous per lane:
//   n = nt*16 + (l&15); k = kt*32 + (l>>4)*8 + (e&7) + (e>>3)*16
// Block offset = ((ntG*Kt + kt) * 512 + l*16 + e), ntG = n0dst/16 + ntLocal.
// ---------------------------------------------------------------------------
__global__ __launch_bounds__(256) void pack_b(const float* __restrict__ W,
                                              __bf16* __restrict__ dst,
                                              int K, int Nsub, int n0dst, int transpose) {
  int tid = blockIdx.x * 256 + threadIdx.x;
  if (tid >= K * Nsub) return;
  int e   = tid & 15;
  int l   = (tid >> 4) & 31;
  int blk = tid >> 9;
  int Kt  = K >> 5;
  int kt  = blk % Kt;
  int ntL = blk / Kt;
  int n   = ntL * 16 + (l & 15);
  int k   = kt * 32 + ((l >> 4) << 3) + (e & 7) + ((e >> 3) << 4);
  float v = transpose ? W[(size_t)n * K + k] : W[(size_t)k * Nsub + n];
  int ntG = (n0dst >> 4) + ntL;
  dst[(((size_t)ntG * Kt + kt) << 9) + l * 16 + e] = (__bf16)v;
}

// ---------------------------------------------------------------------------
// Generic bf16 WMMA GEMM: out[M,N] = A[M,K] * Bpacked + bias
// One wave -> 16x64 output tile (4 x v_wmma_f32_16x16x32_bf16 per K-step).
// 8 waves / block. M%16==0, N%64==0, K%32==0 (all true here).
// ---------------------------------------------------------------------------
__global__ __launch_bounds__(256) void gemm_bf16(const __bf16* __restrict__ A,
                                                 const __bf16* __restrict__ Bp,
                                                 const float* __restrict__ bias,
                                                 __bf16* __restrict__ outB,
                                                 float* __restrict__ outF,
                                                 int M, int N, int K) {
  int lane = threadIdx.x & 31, w = threadIdx.x >> 5;
  int wid    = blockIdx.x * 8 + w;
  int tilesN = N >> 6;
  int wm = wid / tilesN, wn = wid % tilesN;
  int m0 = wm << 4, n0 = wn << 6;
  int Kt = K >> 5;
  int nl = lane & 15, koff = (lane >> 4) * 8;

  const __bf16* arow = A + (size_t)(m0 + nl) * K + koff;

  v8f c0 = {}, c1 = {}, c2 = {}, c3 = {};
  float b0 = bias[n0 + nl];
  float b1 = bias[n0 + 16 + nl];
  float b2 = bias[n0 + 32 + nl];
  float b3 = bias[n0 + 48 + nl];

  for (int kt = 0; kt < Kt; ++kt) {
    v8bf alo = *(const v8bf*)(arow + kt * 32);
    v8bf ahi = *(const v8bf*)(arow + kt * 32 + 16);
    v16bf af = CAT8_8(alo, ahi);

    size_t bb = (size_t)lane * 16;
    v16bf bf0 = *(const v16bf*)(Bp + (((size_t)(wn * 4 + 0) * Kt + kt) << 9) + bb);
    v16bf bf1 = *(const v16bf*)(Bp + (((size_t)(wn * 4 + 1) * Kt + kt) << 9) + bb);
    v16bf bf2 = *(const v16bf*)(Bp + (((size_t)(wn * 4 + 2) * Kt + kt) << 9) + bb);
    v16bf bf3 = *(const v16bf*)(Bp + (((size_t)(wn * 4 + 3) * Kt + kt) << 9) + bb);

    c0 = wmma_bf16(af, bf0, c0);
    c1 = wmma_bf16(af, bf1, c1);
    c2 = wmma_bf16(af, bf2, c2);
    c3 = wmma_bf16(af, bf3, c3);
  }

  int hi = lane >> 4;
#pragma unroll
  for (int r = 0; r < 8; ++r) {
    int row = m0 + r + hi * 8;
    size_t base = (size_t)row * N + n0;
    float v0 = c0[r] + b0;
    float v1 = c1[r] + b1;
    float v2 = c2[r] + b2;
    float v3 = c3[r] + b3;
    if (outB) {
      outB[base + nl]      = (__bf16)v0;
      outB[base + 16 + nl] = (__bf16)v1;
      outB[base + 32 + nl] = (__bf16)v2;
      outB[base + 48 + nl] = (__bf16)v3;
    }
    if (outF) {
      outF[base + nl]      = v0;
      outF[base + 16 + nl] = v1;
      outF[base + 32 + nl] = v2;
      outF[base + 48 + nl] = v3;
    }
  }
}

// ---------------------------------------------------------------------------
// TransformerConv edge attention. qkv layout per node row (2048 cols bf16):
// [0,512)=q  [512,1024)=k  [1024,1536)=v  [1536,2048)=skip
// dst(e) == e/16 by construction. One wave per destination node.
// lane -> head h = lane>>2, 16-dim chunk c = lane&3. Fuses softmax over the
// 16 in-edges, weighted V sum, +skip, ReLU, bf16 output [N, 512].
// ---------------------------------------------------------------------------
__global__ __launch_bounds__(256) void edge_attn(const __bf16* __restrict__ qkv,
                                                 const int* __restrict__ src,
                                                 __bf16* __restrict__ hout) {
  int w = threadIdx.x >> 5, lane = threadIdx.x & 31;
  int node = blockIdx.x * 8 + w;
  int h = lane >> 2, c = lane & 3;
  int cb = h * 64 + c * 16;
  size_t nb = (size_t)node * 2048;

  float qv[16];
  {
    const __bf16* qp = qkv + nb + cb;
    v8bf q0 = *(const v8bf*)qp;
    v8bf q1 = *(const v8bf*)(qp + 8);
#pragma unroll
    for (int i = 0; i < 8; ++i) { qv[i] = (float)q0[i]; qv[8 + i] = (float)q1[i]; }
  }

  int srcs[16];
  const int* sp = src + node * DEG;
#pragma unroll
  for (int e = 0; e < DEG; ++e) srcs[e] = sp[e];

  float logit[16];
  float mx = -1e30f;
#pragma unroll
  for (int e = 0; e < DEG; ++e) {
    const __bf16* kp = qkv + (size_t)srcs[e] * 2048 + 512 + cb;
    v8bf k0 = *(const v8bf*)kp;
    v8bf k1 = *(const v8bf*)(kp + 8);
    float d = 0.f;
#pragma unroll
    for (int i = 0; i < 8; ++i) {
      d = fmaf(qv[i],     (float)k0[i], d);
      d = fmaf(qv[8 + i], (float)k1[i], d);
    }
    d += __shfl_xor(d, 1, 32);
    d += __shfl_xor(d, 2, 32);
    d *= 0.125f;                      // 1/sqrt(HID=64)
    logit[e] = d;
    mx = fmaxf(mx, d);
  }

  float al[16], se = 0.f;
#pragma unroll
  for (int e = 0; e < DEG; ++e) { al[e] = __expf(logit[e] - mx); se += al[e]; }
  float inv = 1.f / se;

  float acc[16];
#pragma unroll
  for (int i = 0; i < 16; ++i) acc[i] = 0.f;
#pragma unroll
  for (int e = 0; e < DEG; ++e) {
    float a = al[e] * inv;
    const __bf16* vp = qkv + (size_t)srcs[e] * 2048 + 1024 + cb;
    v8bf v0 = *(const v8bf*)vp;
    v8bf v1 = *(const v8bf*)(vp + 8);
#pragma unroll
    for (int i = 0; i < 8; ++i) {
      acc[i]     = fmaf(a, (float)v0[i], acc[i]);
      acc[8 + i] = fmaf(a, (float)v1[i], acc[8 + i]);
    }
  }

  const __bf16* skp = qkv + nb + 1536 + cb;
  v8bf s0 = *(const v8bf*)skp;
  v8bf s1 = *(const v8bf*)(skp + 8);
  v8bf o0, o1;
#pragma unroll
  for (int i = 0; i < 8; ++i) {
    o0[i] = (__bf16)fmaxf(acc[i]     + (float)s0[i], 0.f);
    o1[i] = (__bf16)fmaxf(acc[8 + i] + (float)s1[i], 0.f);
  }
  __bf16* op = hout + (size_t)node * HDIM + cb;
  *(v8bf*)op       = o0;
  *(v8bf*)(op + 8) = o1;
}

// ---------------------------------------------------------------------------
// Global MHA. qkvg bf16 [N, 768] (q|k|v of 256 each). One workgroup per
// (graph b, head h); 8 waves, each handles 4 row-tiles of 16 queries.
// Flash-style online softmax over 16 key-chunks of 32, all matmuls via WMMA.
// V staged once in LDS in B-fragment layout; P transposed D->A layout via LDS.
// ---------------------------------------------------------------------------
__global__ __launch_bounds__(256) void mha_kernel(const __bf16* __restrict__ qkvg,
                                                  __bf16* __restrict__ out) {
  int b = blockIdx.x >> 3, h = blockIdx.x & 7;
  __shared__ __align__(16) __bf16 vpack[2][16][32][16]; // 32 KB, B-frag layout
  __shared__ __align__(16) __bf16 aw[8][16][32];        // 8 KB, per-wave P transpose

  int t = threadIdx.x;
  int lane = t & 31, w = t >> 5;
  int nl = lane & 15, hi = lane >> 4, koff = hi * 8;

  // Stage V into LDS in B-fragment layout: V[key k][dim n] = qkvg[b*512+k][512+h*32+n]
  for (int s = t; s < 1024; s += 256) {
    int nt = s >> 9, rem = s & 511;
    int kc = rem >> 5, l = rem & 31;
    int n  = nt * 16 + (l & 15);
    int kb = (l >> 4) * 8;
#pragma unroll
    for (int e = 0; e < 16; ++e) {
      int k = kc * 32 + kb + (e & 7) + ((e >> 3) << 4);
      vpack[nt][kc][l][e] = qkvg[(size_t)(b * NPG + k) * 768 + 512 + h * 32 + n];
    }
  }
  __syncthreads();

  const float scale = 0.17677669529663687f; // 1/sqrt(32)

  for (int rt = w * 4; rt < w * 4 + 4; ++rt) {
    int row0 = rt * 16;
    // Q A-fragment (K-dim = head_dim = 32 -> single fragment)
    const __bf16* qp = qkvg + (size_t)(b * NPG + row0 + nl) * 768 + h * 32 + koff;
    v8bf qlo = *(const v8bf*)qp;
    v8bf qhi = *(const v8bf*)(qp + 16);
    v16bf aq = CAT8_8(qlo, qhi);

    float mrun[8], lrun[8];
#pragma unroll
    for (int r = 0; r < 8; ++r) { mrun[r] = -1e30f; lrun[r] = 0.f; }
    v8f o0 = {}, o1 = {};
    v8f zc = {};

    for (int kc = 0; kc < 16; ++kc) {
      // Scores: B = K^T, and K is row-major -> direct fragment loads
      int key0 = b * NPG + kc * 32 + nl;
      const __bf16* kp0 = qkvg + (size_t)key0 * 768 + 256 + h * 32 + koff;
      v16bf bk0 = CAT8_8(*(const v8bf*)kp0, *(const v8bf*)(kp0 + 16));
      const __bf16* kp1 = qkvg + (size_t)(key0 + 16) * 768 + 256 + h * 32 + koff;
      v16bf bk1 = CAT8_8(*(const v8bf*)kp1, *(const v8bf*)(kp1 + 16));

      v8f s0 = wmma_bf16(aq, bk0, zc);
      v8f s1 = wmma_bf16(aq, bk1, zc);

      // Online softmax update (row r lives in one 16-lane half)
      float psc[8];
#pragma unroll
      for (int r = 0; r < 8; ++r) {
        float a0 = s0[r] * scale, a1 = s1[r] * scale;
        float cm = fmaxf(a0, a1);
        cm = fmaxf(cm, __shfl_xor(cm, 1, 32));
        cm = fmaxf(cm, __shfl_xor(cm, 2, 32));
        cm = fmaxf(cm, __shfl_xor(cm, 4, 32));
        cm = fmaxf(cm, __shfl_xor(cm, 8, 32));
        float mn = fmaxf(mrun[r], cm);
        float sc = __expf(mrun[r] - mn);
        float p0 = __expf(a0 - mn);
        float p1 = __expf(a1 - mn);
        float cs = p0 + p1;
        cs += __shfl_xor(cs, 1, 32);
        cs += __shfl_xor(cs, 2, 32);
        cs += __shfl_xor(cs, 4, 32);
        cs += __shfl_xor(cs, 8, 32);
        lrun[r] = lrun[r] * sc + cs;
        mrun[r] = mn;
        psc[r]  = sc;
        s0[r] = p0;
        s1[r] = p1;
      }
#pragma unroll
      for (int r = 0; r < 8; ++r) { o0[r] *= psc[r]; o1[r] *= psc[r]; }

      // Transpose P from D-layout to A-layout via per-wave LDS buffer
#pragma unroll
      for (int r = 0; r < 8; ++r) {
        aw[w][r + hi * 8][nl]      = (__bf16)s0[r];
        aw[w][r + hi * 8][nl + 16] = (__bf16)s1[r];
      }
      asm volatile("s_wait_dscnt 0" ::: "memory");
      const __bf16* ap = &aw[w][nl][koff];
      v16bf aa = CAT8_8(*(const v8bf*)ap, *(const v8bf*)(ap + 16));

      // O += P * V (V from LDS, B-fragment layout)
      v16bf bv0 = *(const v16bf*)&vpack[0][kc][lane][0];
      v16bf bv1 = *(const v16bf*)&vpack[1][kc][lane][0];
      o0 = wmma_bf16(aa, bv0, o0);
      o1 = wmma_bf16(aa, bv1, o1);
    }

    // Normalize and store bf16 [N, 256]
#pragma unroll
    for (int r = 0; r < 8; ++r) {
      float inv = 1.f / lrun[r];
      size_t grow = (size_t)(b * NPG + row0 + r + hi * 8) * OUTC + h * 32;
      out[grow + nl]      = (__bf16)(o0[r] * inv);
      out[grow + 16 + nl] = (__bf16)(o1[r] * inv);
    }
  }
}

// ---------------------------------------------------------------------------
// LayerNorm over last dim (256), wave per row, 8 waves / block.
// ---------------------------------------------------------------------------
__global__ __launch_bounds__(256) void ln_kernel(const float* __restrict__ x,
                                                 const float* __restrict__ g,
                                                 const float* __restrict__ bta,
                                                 float* __restrict__ out) {
  int w = threadIdx.x >> 5, lane = threadIdx.x & 31;
  int row = blockIdx.x * 8 + w;
  const float* xr = x + (size_t)row * OUTC;
  float v[8];
#pragma unroll
  for (int i = 0; i < 8; ++i) v[i] = xr[lane * 8 + i];
  float s = 0.f;
#pragma unroll
  for (int i = 0; i < 8; ++i) s += v[i];
  s += __shfl_xor(s, 1, 32);  s += __shfl_xor(s, 2, 32);
  s += __shfl_xor(s, 4, 32);  s += __shfl_xor(s, 8, 32);
  s += __shfl_xor(s, 16, 32);
  float mu = s * (1.f / 256.f);
  float vs = 0.f;
#pragma unroll
  for (int i = 0; i < 8; ++i) { float d = v[i] - mu; vs = fmaf(d, d, vs); }
  vs += __shfl_xor(vs, 1, 32);  vs += __shfl_xor(vs, 2, 32);
  vs += __shfl_xor(vs, 4, 32);  vs += __shfl_xor(vs, 8, 32);
  vs += __shfl_xor(vs, 16, 32);
  float inv = rsqrtf(vs * (1.f / 256.f) + 1e-5f);
  float* orow = out + (size_t)row * OUTC;
#pragma unroll
  for (int i = 0; i < 8; ++i) {
    int col = lane * 8 + i;
    orow[col] = (v[i] - mu) * inv * g[col] + bta[col];
  }
}

// ---------------------------------------------------------------------------
// Host launcher
// ---------------------------------------------------------------------------
extern "C" void kernel_launch(void* const* d_in, const int* in_sizes, int n_in,
                              void* d_out, int out_size, void* d_ws, size_t ws_size,
                              hipStream_t stream) {
  (void)in_sizes; (void)n_in; (void)out_size; (void)ws_size;

  const float* x        = (const float*)d_in[0];
  const int*   src      = (const int*)d_in[1];   // edge_index row 0 = src (dst = e/16)
  const float* Wout     = (const float*)d_in[35];
  const float* bout     = (const float*)d_in[36];
  const float* in_w     = (const float*)d_in[37];
  const float* in_b     = (const float*)d_in[38];
  const float* op_w     = (const float*)d_in[39];
  const float* op_b     = (const float*)d_in[40];
  const float* ln_g     = (const float*)d_in[41];
  const float* ln_b     = (const float*)d_in[42];

  // Workspace layout (bytes)
  char* ws = (char*)d_ws;
  size_t off = 0;
  __bf16* hA    = (__bf16*)(ws + off); off += (size_t)NN * HDIM * 2;   // 16.8 MB
  __bf16* hB    = (__bf16*)(ws + off); off += (size_t)NN * HDIM * 2;   // 16.8 MB
  __bf16* qkv   = (__bf16*)(ws + off); off += (size_t)NN * 2048 * 2;   // 64   MB
  __bf16* gout  = (__bf16*)(ws + off); off += (size_t)NN * OUTC * 2;   // 8.4  MB
  __bf16* attnO = (__bf16*)(ws + off); off += (size_t)NN * OUTC * 2;   // 8.4  MB
  float*  oprj  = (float*)(ws + off);  off += (size_t)NN * OUTC * 4;   // 16.8 MB
  __bf16* packB = (__bf16*)(ws + off); off += (size_t)512 * 2048 * 2;  // 2.1  MB
  float*  bcat  = (float*)(ws + off);  off += 2048 * 4;

  // 1) x -> bf16
  cvt_f32_bf16<<<dim3((NN * 128 + 255) / 256), 256, 0, stream>>>(x, hA, NN * 128);

  // 2) 4 TransformerConv layers
  __bf16* hcur = hA;
  __bf16* hnext = hB;
  for (int layer = 0; layer < 4; ++layer) {
    int K = (layer == 0) ? 128 : 512;
    const float* Wq = (const float*)d_in[3 + layer * 8 + 0];
    const float* bq = (const float*)d_in[3 + layer * 8 + 1];
    const float* Wk = (const float*)d_in[3 + layer * 8 + 2];
    const float* bk = (const float*)d_in[3 + layer * 8 + 3];
    const float* Wv = (const float*)d_in[3 + layer * 8 + 4];
    const float* bv = (const float*)d_in[3 + layer * 8 + 5];
    const float* Ws = (const float*)d_in[3 + layer * 8 + 6];
    const float* bs = (const float*)d_in[3 + layer * 8 + 7];

    int pe = K * 512;
    dim3 pg((pe + 255) / 256);
    pack_b<<<pg, 256, 0, stream>>>(Wq, packB, K, 512, 0,    0);
    pack_b<<<pg, 256, 0, stream>>>(Wk, packB, K, 512, 512,  0);
    pack_b<<<pg, 256, 0, stream>>>(Wv, packB, K, 512, 1024, 0);
    pack_b<<<pg, 256, 0, stream>>>(Ws, packB, K, 512, 1536, 0);

    hipMemcpyAsync(bcat + 0,    bq, 512 * 4, hipMemcpyDeviceToDevice, stream);
    hipMemcpyAsync(bcat + 512,  bk, 512 * 4, hipMemcpyDeviceToDevice, stream);
    hipMemcpyAsync(bcat + 1024, bv, 512 * 4, hipMemcpyDeviceToDevice, stream);
    hipMemcpyAsync(bcat + 1536, bs, 512 * 4, hipMemcpyDeviceToDevice, stream);

    // [16384,K] x [K,2048] -> qkv (bf16)
    gemm_bf16<<<dim3(4096), 256, 0, stream>>>(hcur, packB, bcat, qkv, nullptr,
                                              NN, 2048, K);
    // edge attention + skip + relu -> hnext bf16 [N,512]
    edge_attn<<<dim3(NN / 8), 256, 0, stream>>>(qkv, src, hnext);

    __bf16* tmp = hcur; hcur = hnext; hnext = tmp;
  }

  // 3) Wout: [16384,512] x [512,256] -> gout bf16
  pack_b<<<dim3((512 * 256 + 255) / 256), 256, 0, stream>>>(Wout, packB, 512, 256, 0, 0);
  gemm_bf16<<<dim3(512), 256, 0, stream>>>(hcur, packB, bout, gout, nullptr,
                                           NN, 256, 512);

  // 4) in_proj: qkvg = gout @ in_proj_w^T + b  -> [16384,768] bf16
  pack_b<<<dim3((256 * 768 + 255) / 256), 256, 0, stream>>>(in_w, packB, 256, 768, 0, 1);
  gemm_bf16<<<dim3(1536), 256, 0, stream>>>(gout, packB, in_b, qkv, nullptr,
                                            NN, 768, 256);

  // 5) global MHA per (graph, head)
  mha_kernel<<<dim3(NG * 8), 256, 0, stream>>>(qkv, attnO);

  // 6) out_proj: [16384,256] x [256,256] -> f32
  pack_b<<<dim3((256 * 256 + 255) / 256), 256, 0, stream>>>(op_w, packB, 256, 256, 0, 1);
  gemm_bf16<<<dim3(512), 256, 0, stream>>>(attnO, packB, op_b, nullptr, oprj,
                                           NN, 256, 256);

  // 7) LayerNorm -> d_out (f32)
  ln_kernel<<<dim3(NN / 8), 256, 0, stream>>>(oprj, ln_g, ln_b, (float*)d_out);
}
